// SRNN_21036749815898
// MI455X (gfx1250) — matrix operand order repaired and checked by reading, not compiled, not run on hardware
//
#include <hip/hip_runtime.h>
#include <hip/hip_bf16.h>

#define N_IN  256
#define N_REC 1024
#define N_OUTC 20
#define N_B   128
#define N_T   100
#define ALPHA 0.951229424500714f
#define KAPPA 0.951229424500714f
#define THR   1.0f

typedef __attribute__((ext_vector_type(16))) _Float16 v16h;
typedef __attribute__((ext_vector_type(8)))  _Float16 v8h;
typedef __attribute__((ext_vector_type(8)))  float    v8f;

// ---- WMMA f16 tile loaders (wave32, 16x16x32) -----------------------------
// A-matrix 16x32 (MxK): lane L(0..15) row M=L; halves 0..7 = K 0..7, halves
// 8..15 = K 16..23; lanes 16..31 hold K+8 (interleaved split per ISA table).
__device__ __forceinline__ v16h load_a_tile(const _Float16* rowPtr, int kbase, int laneHi) {
    const _Float16* p = rowPtr + kbase + laneHi * 8;
    v8h lo = *reinterpret_cast<const v8h*>(p);
    v8h hi = *reinterpret_cast<const v8h*>(p + 16);
    v16h r;
#pragma unroll
    for (int i = 0; i < 8; ++i) { r[i] = lo[i]; r[8 + i] = hi[i]; }
    return r;
}
// B-matrix 32x16 (KxN): lane L(0..15) column N=L; halves 0..7 = K 0..7, halves
// 8..15 = K 16..23; lanes 16..31 hold K+8 (same interleaved split as A).
__device__ __forceinline__ v16h load_b_tile(const _Float16* rowPtr, int kbase, int laneHi) {
    const _Float16* p = rowPtr + kbase + laneHi * 8;
    v8h lo = *reinterpret_cast<const v8h*>(p);
    v8h hi = *reinterpret_cast<const v8h*>(p + 16);
    v16h r;
#pragma unroll
    for (int i = 0; i < 8; ++i) { r[i] = lo[i]; r[8 + i] = hi[i]; }
    return r;
}

// ---- setup kernels ---------------------------------------------------------
__global__ void k_cvt_wrec(const float* __restrict__ wf, const float* __restrict__ wb,
                           _Float16* __restrict__ of, _Float16* __restrict__ ob) {
    int i = blockIdx.x * blockDim.x + threadIdx.x;
    if (i >= N_REC * N_REC) return;
    int r = i / N_REC, c = i % N_REC;
    float df = (r == c) ? 0.f : wf[i];
    float db = (r == c) ? 0.f : wb[i];
    of[i] = (_Float16)df;
    ob[i] = (_Float16)db;
}

__global__ void k_cvt_f16(const float* __restrict__ in, _Float16* __restrict__ out, int n) {
    int i = blockIdx.x * blockDim.x + threadIdx.x;
    if (i < n) out[i] = (_Float16)in[i];
}

__global__ void k_init_state(_Float16* __restrict__ zf0, _Float16* __restrict__ zb0,
                             float* __restrict__ vf, float* __restrict__ vb) {
    int i = blockIdx.x * blockDim.x + threadIdx.x;
    if (i >= N_B * N_REC) return;
    zf0[i] = (_Float16)0.f; zb0[i] = (_Float16)0.f;
    vf[i] = 0.f; vb[i] = 0.f;
}

// ---- input projection: add[s,b,r] = x[b,t(s)] @ w_in.T + noise[s,b,r] ------
// grid: (R/64, B/16, 2*(T-1)), block 128 (4 waves), one 16x16 tile per wave.
__global__ void k_inproj(const _Float16* __restrict__ xh,
                         const _Float16* __restrict__ winf, const _Float16* __restrict__ winb,
                         const float* __restrict__ noisef, const float* __restrict__ noiseb,
                         float* __restrict__ addf, float* __restrict__ addb) {
    const int lane   = threadIdx.x & 31;
    const int wave   = threadIdx.x >> 5;
    const int laneHi = lane >> 4;
    const int laneLo = lane & 15;
    const int dir    = blockIdx.z / (N_T - 1);
    const int s      = (blockIdx.z % (N_T - 1)) + 1;
    const int t      = dir ? (N_T - s) : (s - 1);
    const int tileN  = blockIdx.x * 4 + wave;
    const int tileM  = blockIdx.y;

    const _Float16* win   = dir ? winb   : winf;
    const float*    noise = dir ? noiseb : noisef;
    float*          add   = dir ? addb   : addf;

    const int brow = tileM * 16 + laneLo;
    const _Float16* arow = xh  + ((size_t)brow * N_T + t) * N_IN;
    const int rcol = tileN * 16 + laneLo;
    const _Float16* wrow = win + (size_t)rcol * N_IN;

    v8f acc = {};
#pragma unroll
    for (int k0 = 0; k0 < N_IN; k0 += 32) {
        v16h a = load_a_tile(arow, k0, laneHi);
        v16h b = load_b_tile(wrow, k0, laneHi);
        acc = __builtin_amdgcn_wmma_f32_16x16x32_f16(false, a, false, b,
                                                     (short)0, acc, false, false);
    }
#pragma unroll
    for (int i = 0; i < 8; ++i) {
        int bb = tileM * 16 + laneHi * 8 + i;   // C layout: lanes16-31 hold M+8
        int rr = tileN * 16 + laneLo;
        size_t idx = ((size_t)s * N_B + bb) * N_REC + rr;
        add[idx] = acc[i] + noise[idx];
    }
}

// ---- one recurrent LIF step (both directions fused) ------------------------
// v_s = ALPHA*v + z_{s-1}@wrec.T + add[s] - z_{s-1}*THR ; z_s = v_s > THR
// grid: (R/64, B/16, 2), block 128 (4 waves). K-loop = 1024/32 WMMAs.
__global__ void k_lifstep(const _Float16* __restrict__ wrecf, const _Float16* __restrict__ wrecb,
                          const float* __restrict__ addf, const float* __restrict__ addb,
                          _Float16* __restrict__ zf, _Float16* __restrict__ zb,
                          float* __restrict__ vf, float* __restrict__ vb, int s) {
    const int lane   = threadIdx.x & 31;
    const int wave   = threadIdx.x >> 5;
    const int laneHi = lane >> 4;
    const int laneLo = lane & 15;
    const int dir    = blockIdx.z;
    const int tileN  = blockIdx.x * 4 + wave;
    const int tileM  = blockIdx.y;

    const _Float16* wrec = dir ? wrecb : wrecf;
    const float*    add  = dir ? addb  : addf;
    _Float16*       z    = dir ? zb    : zf;
    float*          v    = dir ? vb    : vf;

    const _Float16* zprev = z + (size_t)(s - 1) * N_B * N_REC;
    const _Float16* arow  = zprev + (size_t)(tileM * 16 + laneLo) * N_REC;
    const _Float16* wrow  = wrec  + (size_t)(tileN * 16 + laneLo) * N_REC;

    v8f acc = {};
#pragma unroll 4
    for (int k0 = 0; k0 < N_REC; k0 += 32) {
        v16h a = load_a_tile(arow, k0, laneHi);
        v16h b = load_b_tile(wrow, k0, laneHi);
        acc = __builtin_amdgcn_wmma_f32_16x16x32_f16(false, a, false, b,
                                                     (short)0, acc, false, false);
    }
    _Float16* znew = z + (size_t)s * N_B * N_REC;
#pragma unroll
    for (int i = 0; i < 8; ++i) {
        int bb = tileM * 16 + laneHi * 8 + i;
        int rr = tileN * 16 + laneLo;
        size_t vi = (size_t)bb * N_REC + rr;
        float zp = (float)zprev[vi];
        float vn = ALPHA * v[vi] + acc[i] + add[((size_t)s * N_B + bb) * N_REC + rr] - zp * THR;
        v[vi] = vn;
        znew[vi] = (vn > THR) ? (_Float16)1.f : (_Float16)0.f;
    }
}

// ---- output head: u[s,b,o] = z_cat[s,b,:] @ w_out[o,:] ---------------------
__global__ void k_outproj(const _Float16* __restrict__ zf, const _Float16* __restrict__ zb,
                          const float* __restrict__ wout, float* __restrict__ u) {
    int idx = blockIdx.x * blockDim.x + threadIdx.x;
    if (idx >= N_T * N_B * N_OUTC) return;
    int o = idx % N_OUTC;
    int b = (idx / N_OUTC) % N_B;
    int s = idx / (N_OUTC * N_B);
    const _Float16* zr = zf + ((size_t)s * N_B + b) * N_REC;
    const float* w = wout + (size_t)o * (2 * N_REC);
    float acc = 0.f;
#pragma unroll 8
    for (int r = 0; r < N_REC; ++r) acc += (float)zr[r] * w[r];
    int j = N_T - 1 - s;                    // stale-read mask: 1 <= j <= s
    if (j >= 1 && j <= s) {
        const _Float16* zr2 = zb + ((size_t)j * N_B + b) * N_REC;
#pragma unroll 8
        for (int r = 0; r < N_REC; ++r) acc += (float)zr2[r] * w[N_REC + r];
    }
    u[idx] = acc;   // B_O == 0
}

// ---- vo leaky scan: vo[s] = KAPPA*vo[s-1] + u[s], vo[0]=0 ------------------
__global__ void k_voscan(const float* __restrict__ u, float* __restrict__ vo) {
    int idx = blockIdx.x * blockDim.x + threadIdx.x;
    if (idx >= N_B * N_OUTC) return;
    float acc = 0.f;
    vo[idx] = 0.f;
    for (int s = 1; s < N_T; ++s) {
        acc = KAPPA * acc + u[(size_t)s * N_B * N_OUTC + idx];
        vo[(size_t)s * N_B * N_OUTC + idx] = acc;
    }
}

// ---- softmax over O of vo[T,B,O] -> sm[B,T,O] ------------------------------
__global__ void k_softmax(const float* __restrict__ vo, float* __restrict__ sm) {
    int idx = blockIdx.x * blockDim.x + threadIdx.x;
    if (idx >= N_B * N_T) return;
    int s = idx % N_T, b = idx / N_T;
    const float* p = vo + ((size_t)s * N_B + b) * N_OUTC;
    float m = p[0];
#pragma unroll
    for (int o = 1; o < N_OUTC; ++o) m = fmaxf(m, p[o]);
    float e[N_OUTC], sum = 0.f;
#pragma unroll
    for (int o = 0; o < N_OUTC; ++o) { e[o] = __expf(p[o] - m); sum += e[o]; }
    float inv = 1.f / sum;
    float* q = sm + ((size_t)b * N_T + s) * N_OUTC;
#pragma unroll
    for (int o = 0; o < N_OUTC; ++o) q[o] = e[o] * inv;
}

// ---- host-side orchestration ----------------------------------------------
static inline char* carve(char*& p, size_t bytes) {
    char* r = p;
    p += (bytes + 255) & ~(size_t)255;
    return r;
}

extern "C" void kernel_launch(void* const* d_in, const int* in_sizes, int n_in,
                              void* d_out, int out_size, void* d_ws, size_t ws_size,
                              hipStream_t stream) {
    const float* x        = (const float*)d_in[0];
    const float* w_in_fw  = (const float*)d_in[1];
    const float* w_rec_fw = (const float*)d_in[2];
    const float* w_in_bw  = (const float*)d_in[3];
    const float* w_rec_bw = (const float*)d_in[4];
    const float* w_out    = (const float*)d_in[5];
    const float* noise_fw = (const float*)d_in[6];
    const float* noise_bw = (const float*)d_in[7];
    (void)in_sizes; (void)n_in; (void)out_size; (void)ws_size;

    char* p = (char*)d_ws;
    _Float16* wrecf_h = (_Float16*)carve(p, (size_t)N_REC * N_REC * 2);
    _Float16* wrecb_h = (_Float16*)carve(p, (size_t)N_REC * N_REC * 2);
    _Float16* x_h     = (_Float16*)carve(p, (size_t)N_B * N_T * N_IN * 2);
    _Float16* winf_h  = (_Float16*)carve(p, (size_t)N_REC * N_IN * 2);
    _Float16* winb_h  = (_Float16*)carve(p, (size_t)N_REC * N_IN * 2);
    float*    addf    = (float*)   carve(p, (size_t)N_T * N_B * N_REC * 4);
    float*    addb    = (float*)   carve(p, (size_t)N_T * N_B * N_REC * 4);
    _Float16* zf      = (_Float16*)carve(p, (size_t)N_T * N_B * N_REC * 2);
    _Float16* zb      = (_Float16*)carve(p, (size_t)N_T * N_B * N_REC * 2);
    float*    vf      = (float*)   carve(p, (size_t)N_B * N_REC * 4);
    float*    vb      = (float*)   carve(p, (size_t)N_B * N_REC * 4);
    float*    u_out   = (float*)   carve(p, (size_t)N_T * N_B * N_OUTC * 4);

    float* sm_out = (float*)d_out;                       // [B,T,20]
    float* vo_out = (float*)d_out + (size_t)N_B * N_T * N_OUTC; // [T,B,20]

    // setup / conversions
    k_cvt_wrec<<<(N_REC * N_REC + 255) / 256, 256, 0, stream>>>(w_rec_fw, w_rec_bw, wrecf_h, wrecb_h);
    k_cvt_f16<<<((N_B * N_T * N_IN) + 255) / 256, 256, 0, stream>>>(x, x_h, N_B * N_T * N_IN);
    k_cvt_f16<<<((N_REC * N_IN) + 255) / 256, 256, 0, stream>>>(w_in_fw, winf_h, N_REC * N_IN);
    k_cvt_f16<<<((N_REC * N_IN) + 255) / 256, 256, 0, stream>>>(w_in_bw, winb_h, N_REC * N_IN);
    k_init_state<<<(N_B * N_REC + 255) / 256, 256, 0, stream>>>(zf, zb, vf, vb);

    // input projections + noise fold (big parallel WMMA GEMM, both dirs)
    k_inproj<<<dim3(N_REC / 64, N_B / 16, 2 * (N_T - 1)), 128, 0, stream>>>(
        x_h, winf_h, winb_h, noise_fw, noise_bw, addf, addb);

    // serial recurrent chain: 99 steps, both directions per launch
    for (int s = 1; s < N_T; ++s) {
        k_lifstep<<<dim3(N_REC / 64, N_B / 16, 2), 128, 0, stream>>>(
            wrecf_h, wrecb_h, addf, addb, zf, zb, vf, vb, s);
    }

    // output head + scan + softmax
    k_outproj<<<(N_T * N_B * N_OUTC + 255) / 256, 256, 0, stream>>>(zf, zb, w_out, u_out);
    k_voscan<<<(N_B * N_OUTC + 63) / 64, 64, 0, stream>>>(u_out, vo_out);
    k_softmax<<<(N_B * N_T + 127) / 128, 128, 0, stream>>>(vo_out, sm_out);
}
